// MinimumSpanningTree_25795573580019
// MI455X (gfx1250) — matrix operand order repaired and checked by reading, not compiled, not run on hardware
//
#include <hip/hip_runtime.h>
#include <hip/hip_bf16.h>
#include <hip/hip_fp16.h>

// ---------------------------------------------------------------------------
// MinimumSpanningTree pipeline for MI455X (gfx1250, wave32)
//   K1 knorm  : per-node inverse L2 norms (memory-bound pass, 134 MB)
//   K2 kedges : adjacent-pair dot products via v_wmma_f32_16x16x32_f16 Gram
//               diagonals, f16 hi/lo split for ~f32 accuracy -> edge weights.
//               LDS staged TRANSPOSED ([x][k]) so every WMMA fragment is two
//               ds_load_b128 ops (K-contiguous, k = 16*h + s in both A and B).
//   K3 ksort  : per-batch bitonic sort of 8192 (weight,idx) keys in LDS
//   K4 kmstbfs: per-batch sequential Kruskal union-find + reference-faithful
//               BFS (1 thread per batch, 16-way batch parallel)
// ---------------------------------------------------------------------------

typedef __attribute__((ext_vector_type(16))) _Float16 v16h;
typedef __attribute__((ext_vector_type(8)))  _Float16 v8h;
typedef __attribute__((ext_vector_type(8)))  float    v8f;

#define B_    16
#define C_    512
#define H_    64
#define W_    64
#define L_    (H_ * W_)          // 4096 nodes
#define E_ROW ((H_ - 1) * W_)    // 4032 vertical  edges, e = y*64 + x
#define E_COL (H_ * (W_ - 1))    // 4032 horizontal edges, e = 4032 + y*63 + x
#define E_    (E_ROW + E_COL)    // 8064
#define EPAD  8192               // padded to power of two for bitonic sort
#define KP    40                 // padded k-extent (halves): 32 used + 8 pad
                                 // row stride 80 B -> 16B-aligned b128 loads,
                                 // bank starts 20*x mod 64 = conflict-free

// ---------------------------------------------------------------------------
// K1: inverse norms, inv[b][n] = 1 / max(||fm[b,:,n]||, 1e-8)
// ---------------------------------------------------------------------------
__global__ __launch_bounds__(256) void knorm(const float* __restrict__ fm,
                                             float* __restrict__ inv_norm) {
    int g = blockIdx.x * blockDim.x + threadIdx.x;   // over B*L
    int b = g / L_;
    int node = g - b * L_;
    const float* p = fm + (size_t)b * C_ * L_ + node;
    float s = 0.f;
    #pragma unroll 8
    for (int c = 0; c < C_; ++c) {
        float v = p[(size_t)c * L_];                 // coalesced across lanes
        s = fmaf(v, v, s);
    }
    inv_norm[g] = 1.f / fmaxf(sqrtf(s), 1e-8f);
}

// ---------------------------------------------------------------------------
// K2: edge weights. One block per (b, y): 4 waves, wave w owns x0 = 16*w.
// ---------------------------------------------------------------------------
__global__ __launch_bounds__(128) void kedges(const float* __restrict__ fm,
                                              const float* __restrict__ inv_norm,
                                              float* __restrict__ w) {
    // [hi/lo][row y / y+1][x][k]  -- transposed staging, 20 KB
    __shared__ alignas(16) _Float16 sh[2][2][64][KP];

    const int b    = blockIdx.x >> 6;
    const int y    = blockIdx.x & 63;
    const int tid  = threadIdx.x;
    const int lane = tid & 31;
    const int x0   = (tid >> 5) << 4;                // wave id * 16
    const int y1   = (y + 1 < H_) ? (y + 1) : (H_ - 1);
    const size_t bbase = (size_t)b * C_ * L_;

    // loader assignment: 128 threads = (row rr) x (x-quad, c-base), float4 loads
    const int rr = tid >> 6;                         // 0: row y, 1: row y+1
    const int q  = tid & 63;
    const int x4 = (q & 15) << 2;                    // x quad base 0..60
    const int c0 = q >> 4;                           // 0..3
    const int yy = rr ? y1 : y;

    union V16 { v16h v; v8h p[2]; };

    v8f acc_col = {};   // pairs (y, x0+i) -- (y, x0+i+1)
    v8f acc_row = {};   // pairs (y, x0+i) -- (y+1, x0+i)

    for (int k0 = 0; k0 < C_; k0 += 32) {
        // ---- stage 32 k x 64 x (two rows) as hi/lo f16, transposed --------
        #pragma unroll
        for (int i = 0; i < 8; ++i) {
            const int cc = c0 + (i << 2);            // 0..31
            const float* src = fm + bbase + (size_t)(k0 + cc) * L_ + yy * W_ + x4;
            float4 vv = *(const float4*)src;         // global_load_b128
            if (k0 + 32 < C_)                        // prefetch next K-chunk
                __builtin_prefetch(src + (size_t)32 * L_, 0, 0);
            float f[4] = {vv.x, vv.y, vv.z, vv.w};
            #pragma unroll
            for (int t = 0; t < 4; ++t) {
                _Float16 hi = (_Float16)f[t];
                _Float16 lo = (_Float16)(f[t] - (float)hi);
                sh[0][rr][x4 + t][cc] = hi;
                sh[1][rr][x4 + t][cc] = lo;
            }
        }
        __syncthreads();

        // ---- fragments: each = two ds_load_b128 (K-contiguous layout) ----
        V16 a_hi, a_lo, bc_hi, bc_lo, br_hi, br_lo;
        {
            // A 16x32: lane group hi -> halves 0..15 = K 16*hi+h (sequential)
            const int hih8 = (lane >> 4) << 4;       // 0 or 16
            const int am   = x0 + (lane & 15);
            a_hi.p[0] = *(const v8h*)&sh[0][0][am][hih8];
            a_hi.p[1] = *(const v8h*)&sh[0][0][am][8 + hih8];
            a_lo.p[0] = *(const v8h*)&sh[1][0][am][hih8];
            a_lo.p[1] = *(const v8h*)&sh[1][0][am][8 + hih8];
        }
        {
            // B 32x16: lanes 0-15 K=0..15, lanes 16-31 K=16..31 (sequential)
            const int n  = lane & 15;
            const int kg = (lane >> 4) << 4;         // 0 or 16
            const int xc = (x0 + n + 1 < W_) ? (x0 + n + 1) : (W_ - 1);
            const int xr = x0 + n;
            bc_hi.p[0] = *(const v8h*)&sh[0][0][xc][kg];
            bc_hi.p[1] = *(const v8h*)&sh[0][0][xc][kg + 8];
            bc_lo.p[0] = *(const v8h*)&sh[1][0][xc][kg];
            bc_lo.p[1] = *(const v8h*)&sh[1][0][xc][kg + 8];
            br_hi.p[0] = *(const v8h*)&sh[0][1][xr][kg];
            br_hi.p[1] = *(const v8h*)&sh[0][1][xr][kg + 8];
            br_lo.p[0] = *(const v8h*)&sh[1][1][xr][kg];
            br_lo.p[1] = *(const v8h*)&sh[1][1][xr][kg + 8];
        }

        acc_col = __builtin_amdgcn_wmma_f32_16x16x32_f16(false, a_hi.v, false, bc_hi.v, (short)0, acc_col, false, false);
        acc_col = __builtin_amdgcn_wmma_f32_16x16x32_f16(false, a_hi.v, false, bc_lo.v, (short)0, acc_col, false, false);
        acc_col = __builtin_amdgcn_wmma_f32_16x16x32_f16(false, a_lo.v, false, bc_hi.v, (short)0, acc_col, false, false);

        acc_row = __builtin_amdgcn_wmma_f32_16x16x32_f16(false, a_hi.v, false, br_hi.v, (short)0, acc_row, false, false);
        acc_row = __builtin_amdgcn_wmma_f32_16x16x32_f16(false, a_hi.v, false, br_lo.v, (short)0, acc_row, false, false);
        acc_row = __builtin_amdgcn_wmma_f32_16x16x32_f16(false, a_lo.v, false, br_hi.v, (short)0, acc_row, false, false);

        __syncthreads();
    }

    // ---- extract diagonal: f32 C/D layout -------------------------------
    // lanes 0-15 hold M = r (VGPR r), N = lane; lanes 16-31 hold M = 8+r, N = lane-16
    float dcol = 0.f, drow = 0.f;
    #pragma unroll
    for (int r = 0; r < 8; ++r) {
        if (lane == r)      { dcol = acc_col[r]; drow = acc_row[r]; }   // i = r
        if (lane == 24 + r) { dcol = acc_col[r]; drow = acc_row[r]; }   // i = 8+r
    }
    const bool have = (lane < 8) || (lane >= 24);
    const int  i    = (lane < 16) ? lane : (lane - 16);

    if (have) {
        const int xi = x0 + i;
        const float* nrm = inv_norm + (size_t)b * L_;
        const float nu = nrm[y * W_ + xi];
        if (xi < W_ - 1) {   // horizontal edge (y,xi)-(y,xi+1)
            float sim = dcol * nu * nrm[y * W_ + xi + 1];
            w[(size_t)b * EPAD + E_ROW + y * (W_ - 1) + xi] = -sim;
        }
        if (y < H_ - 1) {    // vertical edge (y,xi)-(y+1,xi)
            float sim = drow * nu * nrm[(y + 1) * W_ + xi];
            w[(size_t)b * EPAD + y * W_ + xi] = -sim;
        }
    }
}

// ---------------------------------------------------------------------------
// K3: per-batch bitonic sort of 8192 keys in LDS (64 KB of 320 KB WGP LDS).
// Key = (sortable-float(weight) << 32) | edge_index  -> stable-by-index ties.
// ---------------------------------------------------------------------------
__global__ __launch_bounds__(1024) void ksort(const float* __restrict__ w,
                                              unsigned int* __restrict__ sorted_idx) {
    __shared__ unsigned long long keys[EPAD];
    const int b   = blockIdx.x;
    const int tid = threadIdx.x;

    for (int i = tid; i < EPAD; i += blockDim.x) {
        unsigned long long key;
        if (i < E_) {
            unsigned int u = __float_as_uint(w[(size_t)b * EPAD + i]);
            u = (u & 0x80000000u) ? ~u : (u | 0x80000000u);   // total order
            key = ((unsigned long long)u << 32) | (unsigned int)i;
        } else {
            key = ~0ull;                                      // pad to the end
        }
        keys[i] = key;
    }
    __syncthreads();

    for (int k = 2; k <= EPAD; k <<= 1) {
        for (int j = k >> 1; j > 0; j >>= 1) {
            for (int i = tid; i < EPAD; i += blockDim.x) {
                int partner = i ^ j;
                if (partner > i) {
                    bool up = ((i & k) == 0);
                    unsigned long long a = keys[i], c = keys[partner];
                    if ((a > c) == up) { keys[i] = c; keys[partner] = a; }
                }
            }
            __syncthreads();
        }
    }

    for (int i = tid; i < EPAD; i += blockDim.x)
        sorted_idx[(size_t)b * EPAD + i] = (unsigned int)(keys[i] & 0xFFFFFFFFu);
}

// ---------------------------------------------------------------------------
// K4: sequential Kruskal (path-halving union-find, union-by-rank) + BFS,
// one thread per batch. Faithful to reference: node_to_pos assigned at
// enqueue time = #popped-so-far; neighbors visited in sorted order.
// Output written as float (all values exact in f32).
// ---------------------------------------------------------------------------
__global__ __launch_bounds__(32) void kmstbfs(const unsigned int* __restrict__ sorted_idx,
                                              int* __restrict__ scratch,
                                              float* __restrict__ out) {
    const int b = blockIdx.x;
    if (threadIdx.x != 0) return;

    int* parent = scratch + (size_t)b * (11 * L_);
    int* rnk    = parent + L_;
    int* adj    = rnk + L_;          // L_ * 4
    int* deg    = adj + 4 * L_;
    int* npos   = deg + L_;
    int* ppos   = npos + L_;
    int* vis    = ppos + L_;
    int* que    = vis + L_;

    for (int i = 0; i < L_; ++i) {
        parent[i] = i; rnk[i] = 0; deg[i] = 0; vis[i] = 0; ppos[i] = -1;
    }

    // Kruskal
    int cnt = 0;
    for (int e = 0; e < E_ && cnt < L_ - 1; ++e) {
        int ei = (int)sorted_idx[(size_t)b * EPAD + e];
        int u, v;
        if (ei < E_ROW) { u = ei; v = ei + W_; }
        else { int t = ei - E_ROW; int yy = t / (W_ - 1); int xx = t - yy * (W_ - 1);
               u = yy * W_ + xx; v = u + 1; }
        int pu = u; while (parent[pu] != pu) { parent[pu] = parent[parent[pu]]; pu = parent[pu]; }
        int pv = v; while (parent[pv] != pv) { parent[pv] = parent[parent[pv]]; pv = parent[pv]; }
        if (pu == pv) continue;
        if (rnk[pu] < rnk[pv]) { int t = pu; pu = pv; pv = t; }
        parent[pv] = pu;
        if (rnk[pu] == rnk[pv]) rnk[pu]++;
        adj[u * 4 + deg[u]++] = v;
        adj[v * 4 + deg[v]++] = u;
        cnt++;
    }

    // BFS from node 0
    float* out_idx = out + (size_t)b * L_;
    float* out_par = out + (size_t)B_ * L_ + (size_t)b * L_;
    int head = 0, tail = 0, popped = 0;
    vis[0] = 1; npos[0] = 0; que[tail++] = 0;
    while (head < tail) {
        int node = que[head++];
        out_idx[popped] = (float)node;
        out_par[popped] = (float)ppos[node];
        popped++;                                   // = len(bfs_order)
        int nb[4]; int d = deg[node];
        for (int t = 0; t < d; ++t) nb[t] = adj[node * 4 + t];
        for (int a2 = 1; a2 < d; ++a2) {            // sorted(adj[node])
            int key = nb[a2]; int b2 = a2 - 1;
            while (b2 >= 0 && nb[b2] > key) { nb[b2 + 1] = nb[b2]; b2--; }
            nb[b2 + 1] = key;
        }
        for (int t = 0; t < d; ++t) {
            int x = nb[t];
            if (!vis[x]) { vis[x] = 1; npos[x] = popped; ppos[x] = npos[node]; que[tail++] = x; }
        }
    }
}

// ---------------------------------------------------------------------------
extern "C" void kernel_launch(void* const* d_in, const int* in_sizes, int n_in,
                              void* d_out, int out_size, void* d_ws, size_t ws_size,
                              hipStream_t stream) {
    (void)in_sizes; (void)n_in; (void)out_size; (void)ws_size;
    const float* fm = (const float*)d_in[0];
    float* out = (float*)d_out;

    // workspace layout (~4.2 MB total)
    float* inv_norm      = (float*)d_ws;                              // B*L f32
    float* w             = inv_norm + (size_t)B_ * L_;                // B*EPAD f32
    unsigned int* sorted = (unsigned int*)(w + (size_t)B_ * EPAD);    // B*EPAD u32
    int* scratch         = (int*)(sorted + (size_t)B_ * EPAD);        // B*11*L i32

    knorm  <<<(B_ * L_) / 256, 256, 0, stream>>>(fm, inv_norm);
    kedges <<<B_ * H_,         128, 0, stream>>>(fm, inv_norm, w);
    ksort  <<<B_,             1024, 0, stream>>>(w, sorted);
    kmstbfs<<<B_,               32, 0, stream>>>(sorted, scratch, out);
}
